// MultiHeadAttention_50878182588627
// MI455X (gfx1250) — compile-verified, matrix operand order checked by reading, not run on hardware
//
#include <hip/hip_runtime.h>
#include <hip/hip_bf16.h>

#define B_  4
#define S_  2048
#define D_  1024
#define H_  16
#define HD_ 64
#define LOG2E 1.4426950408889634f

typedef __attribute__((ext_vector_type(16))) __bf16 v16bf;
typedef __attribute__((ext_vector_type(8)))  float  v8f;
typedef __attribute__((ext_vector_type(8)))  __bf16 bf16x8;
typedef __attribute__((ext_vector_type(4)))  unsigned int u32x4;
typedef __attribute__((ext_vector_type(8)))  int i32x8;
typedef __attribute__((ext_vector_type(4)))  int i32x4;

static __device__ __forceinline__ v8f wmma_bf16f32(v16bf a, v16bf b, v8f c) {
  // (neg_a, A, neg_b, B, c_mod, C, reuse_a, reuse_b)
  return __builtin_amdgcn_wmma_f32_16x16x32_bf16(false, a, false, b, (short)0, c, false, false);
}

// Load 8 contiguous bf16 (16B aligned) into halves [h0, h0+8) of a fragment.
static __device__ __forceinline__ void load8(v16bf& f, int h0, const __bf16* p) {
  bf16x8 v = *(const bf16x8*)p;
#pragma unroll
  for (int i = 0; i < 8; ++i) f[h0 + i] = v[i];
}

// ---------------------------------------------------------------------------
// Tensor Data Mover: 2-D tile load Global -> LDS (bf16 elements).
//   lds_off      : byte offset into the workgroup's dynamic LDS
//   gaddr        : global address of tile origin
//   tile_d0      : contiguous elements per row
//   tile_d1      : number of rows
//   stride_elems : row stride in the global tensor (elements)
// D# layout per CDNA5 ISA ch.8 (group0: count/lds/global/type; group1:
// data_size, tensor dims, tile dims, dim0 stride). Groups 2/3 zero (2-D).
// ---------------------------------------------------------------------------
static __device__ __forceinline__ void tdm_load_2d(unsigned int lds_off,
                                                   const void* gaddr,
                                                   unsigned int tile_d0,
                                                   unsigned int tile_d1,
                                                   unsigned int stride_elems) {
  unsigned long long ga = (unsigned long long)(uintptr_t)gaddr;
  u32x4 g0;
  g0[0] = 1u;                                             // count=1 (valid user D#)
  g0[1] = lds_off;                                        // lds_addr
  g0[2] = (unsigned int)ga;                               // global_addr[31:0]
  g0[3] = (unsigned int)((ga >> 32) & 0x1FFFFFFu) | (2u << 30);  // addr[56:32] | type=2
  i32x8 g1;
  g1[0] = (int)(1u << 16);                                // wg_mask=0, data_size=1 (2B)
  g1[1] = (int)((tile_d0 & 0xFFFFu) << 16);               // tensor_dim0 = tile_d0 (lo16)
  g1[2] = (int)(((tile_d0 >> 16) & 0xFFFFu) |             // tensor_dim0 hi16
                ((tile_d1 & 0xFFFFu) << 16));             // tensor_dim1 = tile_d1 (lo16)
  g1[3] = (int)(((tile_d1 >> 16) & 0xFFFFu) |             // tensor_dim1 hi16
                ((tile_d0 & 0xFFFFu) << 16));             // tile_dim0
  g1[4] = (int)(tile_d1 & 0xFFFFu);                       // tile_dim1 (tile_dim2=0)
  g1[5] = (int)stride_elems;                              // tensor_dim0_stride lo32
  g1[6] = 0;                                              // stride hi16 | dim1_stride lo16
  g1[7] = 0;                                              // dim1_stride hi32
  i32x4 z4 = {0, 0, 0, 0};
#if __clang_major__ >= 23
  i32x8 z8 = {0, 0, 0, 0, 0, 0, 0, 0};
  __builtin_amdgcn_tensor_load_to_lds(g0, g1, z4, z4, z8, 0);
#else
  __builtin_amdgcn_tensor_load_to_lds(g0, g1, z4, z4, 0);
#endif
}

// ---------------------------------------------------------------------------
// fp32 -> bf16 conversion
// ---------------------------------------------------------------------------
__global__ void cvt_f32_to_bf16(const float* __restrict__ in,
                                __bf16* __restrict__ out, int n) {
  int i = (blockIdx.x * blockDim.x + threadIdx.x) * 4;
  if (i + 3 < n) {
    __builtin_prefetch(in + i + 4096, 0, 1);    // global_prefetch_b8 next chunk
    float4 v = *(const float4*)(in + i);
    out[i + 0] = (__bf16)v.x;
    out[i + 1] = (__bf16)v.y;
    out[i + 2] = (__bf16)v.z;
    out[i + 3] = (__bf16)v.w;
  } else {
    for (; i < n; ++i) out[i] = (__bf16)in[i];
  }
}

// ---------------------------------------------------------------------------
// C[M,N] = A[M,K] @ W[N,K]^T.
// Both 64x32 panels (A rows of this block, W rows of this block) are streamed
// into LDS by the Tensor Data Mover, double-buffered: wave 0 issues the DMA
// for panel k+32, all waves run WMMAs on panel k from LDS (ds_load_b128
// fragments, no register rotation), then TENSORcnt wait + barrier flips the
// buffer. W panel is fetched once per block instead of once per wave.
// LDS: [0,8K) A tiles (2 x 64x32 bf16), [8K,16K) W tiles (2 x 64x32 bf16).
// mode 0: bf16 out [M,N] | mode 1: bf16 out scattered to VT [B,H,HD,S]
// mode 2: fp32 out [M,N] + bias[N]
// ---------------------------------------------------------------------------
__global__ void gemm_wmma_bf16(const __bf16* __restrict__ A,
                               const __bf16* __restrict__ W,
                               __bf16* __restrict__ outb,
                               float* __restrict__ outf,
                               const float* __restrict__ bias,
                               int M, int N, int Kd, int mode) {
  extern __shared__ char smem[];
  const unsigned int ABUF = 0, WBUF = 8192;

  const int lane = threadIdx.x & 31;
  const int wave = threadIdx.x >> 5;
  const int ln16 = lane & 15;
  const int hihf = lane >> 4;
  const int kl   = hihf << 3;               // per-lane K sub-offset {0,8}
  const int m0   = blockIdx.x * 64 + wave * 16;
  const int n0   = blockIdx.y * 64;

  const __bf16* aB = A + (size_t)(blockIdx.x * 64) * Kd;   // block's 64 A rows
  const __bf16* wB = W + (size_t)n0 * Kd;                  // block's 64 W rows

  v8f acc[4] = {};

  if (wave == 0) {                          // prologue DMA: panel kb=0
    tdm_load_2d(ABUF, aB, 32, 64, Kd);
    tdm_load_2d(WBUF, wB, 32, 64, Kd);
    __builtin_amdgcn_s_wait_tensorcnt(0);
  }
  __syncthreads();

  for (int kb = 0; kb < Kd; kb += 32) {
    const int cur = (kb >> 5) & 1;
    const bool more = (kb + 32 < Kd);
    if (more && wave == 0) {                // DMA next panel while we compute
      tdm_load_2d(ABUF + (1 - cur) * 4096, aB + kb + 32, 32, 64, Kd);
      tdm_load_2d(WBUF + (1 - cur) * 4096, wB + kb + 32, 32, 64, Kd);
    }

    const __bf16* at = (const __bf16*)(smem + ABUF + cur * 4096);  // [64][32]
    const __bf16* wt = (const __bf16*)(smem + WBUF + cur * 4096);  // [64][32]

    v16bf a;
    const __bf16* ar = at + (size_t)(wave * 16 + ln16) * 32;
    load8(a, 0, ar + kl);
    load8(a, 8, ar + 16 + kl);
#pragma unroll
    for (int nt = 0; nt < 4; ++nt) {
      const __bf16* wr = wt + (size_t)(nt * 16 + ln16) * 32;
      v16bf wf;
      load8(wf, 0, wr + kl);
      load8(wf, 8, wr + 16 + kl);
      acc[nt] = wmma_bf16f32(a, wf, acc[nt]);
    }

    __syncthreads();                        // everyone done reading cur buffer
    if (more) {
      if (wave == 0) __builtin_amdgcn_s_wait_tensorcnt(0);
      __syncthreads();                      // next panel visible to all waves
    }
  }

#pragma unroll
  for (int nt = 0; nt < 4; ++nt) {
#pragma unroll
    for (int r = 0; r < 8; ++r) {
      const int m = m0 + r + 8 * hihf;      // C layout: VGPR r -> M=r / r+8
      const int n = n0 + nt * 16 + ln16;    //           lane   -> N
      const float v = acc[nt][r];
      if (mode == 0) {
        outb[(size_t)m * N + n] = (__bf16)v;
      } else if (mode == 1) {
        const int bb = m >> 11, s = m & (S_ - 1);
        const int hh = n >> 6, hd = n & 63;
        outb[((size_t)(bb * H_ + hh) * HD_ + hd) * S_ + s] = (__bf16)v;
      } else {
        outf[(size_t)m * N + n] = v + bias[n];
      }
    }
  }
}

// ---------------------------------------------------------------------------
// Causal flash attention. K/V tiles staged into LDS by the Tensor Data Mover
// (double-buffered, wave 0 issues, s_wait_tensorcnt + barrier), so the 4 waves
// share one copy of each 32-key tile instead of 4 redundant global gathers.
// Dynamic LDS: [0,8K) K tiles (2 x 32x64 bf16), [8K,16K) V tiles (2 x 64x32),
// [16K,20K) per-wave P staging.
// ---------------------------------------------------------------------------
__global__ void flash_attn_bf16(const __bf16* __restrict__ Q,
                                const __bf16* __restrict__ Kt,
                                const __bf16* __restrict__ VT,
                                __bf16* __restrict__ CTX) {
  extern __shared__ char smem[];
  const unsigned int KBUF = 0, VBUF = 8192, PBUF = 16384;

  const int lane = threadIdx.x & 31;
  const int wave = threadIdx.x >> 5;
  const int ln16 = lane & 15;
  const int hihf = lane >> 4;
  const int kl   = hihf << 3;

  const int qt = blockIdx.x % (S_ / 64);
  const int bh = blockIdx.x / (S_ / 64);
  const int h  = bh % H_;
  const int b  = bh / H_;
  const int qbase = qt * 64 + wave * 16;

  // Q fragments (16 rows x HD=64) stay in registers for the whole kernel.
  const __bf16* qp = Q + ((size_t)b * S_ + (qbase + ln16)) * D_ + h * HD_;
  v16bf aQ[2];
#pragma unroll
  for (int ks = 0; ks < 2; ++ks) {
    load8(aQ[ks], 0, qp + ks * 32 + kl);
    load8(aQ[ks], 8, qp + ks * 32 + 16 + kl);
  }

  float mr[8], ls[8];
#pragma unroll
  for (int r = 0; r < 8; ++r) { mr[r] = -1e30f; ls[r] = 0.0f; }
  v8f acc[4] = {};

  const __bf16* kgbase = Kt + (size_t)b * S_ * D_ + h * HD_;           // +s*D_
  const __bf16* vgbase = VT + (size_t)(b * H_ + h) * HD_ * S_;         // +hd*S_+s
  __bf16* pb = (__bf16*)(smem + PBUF + wave * 1024);                   // [16][32]

  const int ntiles = qt * 2 + 2;            // block-uniform causal tile count

  if (wave == 0) {                          // prologue DMA: tile 0
    tdm_load_2d(KBUF, kgbase, 64, 32, D_);
    tdm_load_2d(VBUF, vgbase, 32, 64, S_);
    __builtin_amdgcn_s_wait_tensorcnt(0);
  }
  __syncthreads();

  for (int t = 0; t < ntiles; ++t) {
    const int kb  = t * 32;
    const int cur = t & 1;

    if ((t + 1 < ntiles) && wave == 0) {    // prefetch next tile via TDM
      tdm_load_2d(KBUF + (1 - cur) * 4096, kgbase + (size_t)(kb + 32) * D_, 64, 32, D_);
      tdm_load_2d(VBUF + (1 - cur) * 4096, vgbase + (kb + 32), 32, 64, S_);
    }

    if (kb < qbase + 16) {                  // wave-uniform causal predicate
      const __bf16* kt = (const __bf16*)(smem + KBUF + cur * 4096);    // [32][64]
      const __bf16* vt = (const __bf16*)(smem + VBUF + cur * 4096);    // [64][32]

      // ---- scores: two 16x16 tiles = Q(16x64) @ K^T(64x32), K from LDS ----
      v8f sc[2];
#pragma unroll
      for (int nt = 0; nt < 2; ++nt) {
        const __bf16* kr = kt + (size_t)(nt * 16 + ln16) * 64;
        v16bf bk0, bk1;
        load8(bk0, 0, kr + kl);       load8(bk0, 8, kr + 16 + kl);
        load8(bk1, 0, kr + 32 + kl);  load8(bk1, 8, kr + 48 + kl);
        v8f c = {};
        c = wmma_bf16f32(aQ[0], bk0, c);
        c = wmma_bf16f32(aQ[1], bk1, c);
        sc[nt] = c;
      }

      // ---- mask + scale + online softmax (16-lane row reductions) ----
#pragma unroll
      for (int r = 0; r < 8; ++r) {
        const int row = qbase + r + 8 * hihf;
        float s0 = (kb + ln16      <= row) ? sc[0][r] * 0.125f : -1e30f;
        float s1 = (kb + 16 + ln16 <= row) ? sc[1][r] * 0.125f : -1e30f;
        float tm = fmaxf(s0, s1);
#pragma unroll
        for (int off = 1; off < 16; off <<= 1) tm = fmaxf(tm, __shfl_xor(tm, off, 16));
        const float mnew = fmaxf(mr[r], tm);
        const float al = exp2f((mr[r] - mnew) * LOG2E);
        const float p0 = exp2f((s0 - mnew) * LOG2E);
        const float p1 = exp2f((s1 - mnew) * LOG2E);
        float rs = p0 + p1;
#pragma unroll
        for (int off = 1; off < 16; off <<= 1) rs += __shfl_xor(rs, off, 16);
        ls[r] = ls[r] * al + rs;
        mr[r] = mnew;
#pragma unroll
        for (int nt = 0; nt < 4; ++nt) acc[nt][r] *= al;
        sc[0][r] = p0;
        sc[1][r] = p1;
      }

      // ---- P (f32 C-layout) -> per-wave LDS -> A-fragment (16x32 bf16) ----
#pragma unroll
      for (int nt = 0; nt < 2; ++nt)
#pragma unroll
        for (int r = 0; r < 8; ++r)
          pb[(r + 8 * hihf) * 32 + nt * 16 + ln16] = (__bf16)sc[nt][r];

      v16bf aP;                              // LDS is in-order within a wave
      load8(aP, 0, pb + ln16 * 32 + kl);
      load8(aP, 8, pb + ln16 * 32 + 16 + kl);

      // ---- ctx += P(16x32) @ V(32x64), V from LDS (VT rows = hd) ----
#pragma unroll
      for (int nt = 0; nt < 4; ++nt) {
        const __bf16* vr = vt + (size_t)(nt * 16 + ln16) * 32;
        v16bf bv;
        load8(bv, 0, vr + kl);
        load8(bv, 8, vr + 16 + kl);
        acc[nt] = wmma_bf16f32(aP, bv, acc[nt]);
      }
    }

    __syncthreads();                         // everyone done reading buffers
    if (t + 1 < ntiles) {
      if (wave == 0) __builtin_amdgcn_s_wait_tensorcnt(0);
      __syncthreads();                       // next tile visible to all waves
    }
  }

  // ---- normalize and store ctx in [B,S,H,HD] bf16 ----
#pragma unroll
  for (int r = 0; r < 8; ++r) {
    const float inv = 1.0f / ls[r];
    const size_t orow = ((size_t)b * S_ + (qbase + r + 8 * hihf)) * D_ + h * HD_;
#pragma unroll
    for (int nt = 0; nt < 4; ++nt)
      CTX[orow + nt * 16 + ln16] = (__bf16)(acc[nt][r] * inv);
  }
}

// ---------------------------------------------------------------------------
extern "C" void kernel_launch(void* const* d_in, const int* in_sizes, int n_in,
                              void* d_out, int out_size, void* d_ws, size_t ws_size,
                              hipStream_t stream) {
  (void)in_sizes; (void)n_in; (void)out_size; (void)ws_size;
  const float* x   = (const float*)d_in[0];
  const float* wq  = (const float*)d_in[1];
  const float* wk  = (const float*)d_in[2];
  const float* wv  = (const float*)d_in[3];
  const float* wo  = (const float*)d_in[4];
  const float* wob = (const float*)d_in[5];

  const size_t MX = (size_t)B_ * S_;     // 8192
  const size_t EX = MX * D_;             // 8,388,608 activations
  const size_t EW = (size_t)D_ * D_;     // 1,048,576 per weight

  // Workspace layout (bf16 elements), ~88 MB total.
  __bf16* Xb   = (__bf16*)d_ws;
  __bf16* Wqb  = Xb + EX;
  __bf16* Wkb  = Wqb + EW;
  __bf16* Wvb  = Wkb + EW;
  __bf16* Wob  = Wvb + EW;
  __bf16* Qb   = Wob + EW;
  __bf16* Kb   = Qb + EX;
  __bf16* VTb  = Kb + EX;
  __bf16* CTXb = VTb + EX;

  auto cvt = [&](const float* src, __bf16* dst, size_t n) {
    int nblk = (int)((n / 4 + 255) / 256);
    cvt_f32_to_bf16<<<nblk, 256, 0, stream>>>(src, dst, (int)n);
  };
  cvt(x,  Xb,  EX);
  cvt(wq, Wqb, EW);
  cvt(wk, Wkb, EW);
  cvt(wv, Wvb, EW);
  cvt(wo, Wob, EW);

  dim3 ggrid((unsigned)(MX / 64), (unsigned)(D_ / 64));   // 128 x 16
  gemm_wmma_bf16<<<ggrid, 128, 16384, stream>>>(Xb, Wqb, Qb,  nullptr, nullptr,
                                                (int)MX, D_, D_, 0);
  gemm_wmma_bf16<<<ggrid, 128, 16384, stream>>>(Xb, Wkb, Kb,  nullptr, nullptr,
                                                (int)MX, D_, D_, 0);
  gemm_wmma_bf16<<<ggrid, 128, 16384, stream>>>(Xb, Wvb, VTb, nullptr, nullptr,
                                                (int)MX, D_, D_, 1);

  flash_attn_bf16<<<B_ * H_ * (S_ / 64), 128, 20480, stream>>>(Qb, Kb, VTb, CTXb);

  gemm_wmma_bf16<<<ggrid, 128, 16384, stream>>>(CTXb, Wob, nullptr, (float*)d_out, wob,
                                                (int)MX, D_, D_, 2);
}
